// GNNModule_49752901157443
// MI455X (gfx1250) — matrix operand order repaired and checked by reading, not compiled, not run on hardware
//
#include <hip/hip_runtime.h>

typedef _Float16 h16;
typedef __attribute__((ext_vector_type(4)))  _Float16 v4h;
typedef __attribute__((ext_vector_type(8)))  _Float16 v8h;
typedef __attribute__((ext_vector_type(16))) _Float16 v16h;
typedef __attribute__((ext_vector_type(8)))  float    v8f;

#define EE 8192
#define NN 4096
#define LDSP 72   // padded f16 stride: 144B rows -> conflict-free b128 lane fetches

union V16U { v16h v; v8h h[2]; };

// Generic pointer -> 32-bit LDS (workgroup-relative) address for async-to-LDS ops.
__device__ __forceinline__ unsigned lds_addr_of(const void* p) {
    return (unsigned)(uintptr_t)(__attribute__((address_space(3))) const void*)p;
}

// ---------------------------------------------------------------------------
// C[M x 64] = A[M x K] * B[K x 64], B given transposed+f16: BT[n*K + k].
// ATRANS=0: A row-major, ld = K.  ATRANS=1: A[m][k] = Asrc[k*ldsrc + m].
// WG = 256 thr = 8 waves; each WG owns a 64x64 C block; K staged in 64-chunks.
// A: f32 global -> v_cvt_pk_f16_f32 -> LDS.  B: CDNA5 async global->LDS copy.
// ---------------------------------------------------------------------------
template <bool ATRANS>
__global__ __launch_bounds__(256) void gemm_wmma_kernel(
    const float* __restrict__ A, const h16* __restrict__ BT,
    float* __restrict__ C, int M, int K, int ldsrc)
{
    __shared__ h16 lA[64 * LDSP];
    __shared__ h16 lB[64 * LDSP];
    const int t    = threadIdx.x;
    const int m0   = blockIdx.x * 64;
    const int lane = t & 31;
    const int w    = t >> 5;
    const int lsub = lane & 15;
    const int lhi  = lane >> 4;
    const int mtile = w >> 1;          // 0..3
    const int ntb   = (w & 1) * 2;     // n-tiles {0,1} or {2,3}

    v8f acc0 = {}; v8f acc1 = {};

    const int row = t >> 2;            // 0..63 (also B column for staging)
    const int seg = t & 3;             // 16-element segment of the 64-K chunk

    for (int k0 = 0; k0 < K; k0 += 64) {
        __syncthreads();               // protect previous chunk's LDS reads
        // ---- B chunk: async f16 copy global -> LDS (ASYNCcnt path) -------
        {
            const h16* src = BT + (size_t)row * K + k0 + seg * 16;
            unsigned dstOff = lds_addr_of(lB + row * LDSP + seg * 16);
            unsigned long long ga = (unsigned long long)(uintptr_t)src;
            asm volatile("global_load_async_to_lds_b128 %0, %1, off"
                         :: "v"(dstOff), "v"(ga) : "memory");
            asm volatile("global_load_async_to_lds_b128 %0, %1, off offset:16"
                         :: "v"(dstOff), "v"(ga) : "memory");
        }
        // ---- A chunk: f32 -> f16 convert-stage into LDS ------------------
        if (!ATRANS) {
            const float* src = A + (size_t)(m0 + row) * K + k0 + seg * 16;
            h16* dst = lA + row * LDSP + seg * 16;
#pragma unroll
            for (int i = 0; i < 4; ++i) {
                float4 f = ((const float4*)src)[i];
                v4h hv = { (h16)f.x, (h16)f.y, (h16)f.z, (h16)f.w };
                *(v4h*)(dst + 4 * i) = hv;
            }
            if (k0 + 64 < K) __builtin_prefetch(src + 64, 0, 3); // global_prefetch_b8
        } else {
            const int mm = t & 63, kk = t >> 6;
#pragma unroll 4
            for (int p = 0; p < 16; ++p) {
                int k = p * 4 + kk;
                float f = A[(size_t)(k0 + k) * ldsrc + m0 + mm];
                lA[mm * LDSP + k] = (h16)f;
            }
        }
        asm volatile("s_wait_asynccnt 0" ::: "memory");
        __syncthreads();
        // ---- two 16x16x32 WMMA steps per chunk ---------------------------
#pragma unroll
        for (int ks = 0; ks < 2; ++ks) {
            const int kA = ks * 32;
            V16U a;
            const h16* pa = lA + (mtile * 16 + lsub) * LDSP + kA + 16 * lhi;
            a.h[0] = *(const v8h*)(pa);          // K = kA + 16*lhi + 0..7
            a.h[1] = *(const v8h*)(pa + 8);      // K = kA + 16*lhi + 8..15
            V16U b0, b1;
            const h16* pb0 = lB + ((ntb    ) * 16 + lsub) * LDSP + kA + 16 * lhi;
            const h16* pb1 = lB + ((ntb + 1) * 16 + lsub) * LDSP + kA + 16 * lhi;
            b0.h[0] = *(const v8h*)(pb0); b0.h[1] = *(const v8h*)(pb0 + 8);
            b1.h[0] = *(const v8h*)(pb1); b1.h[1] = *(const v8h*)(pb1 + 8);
            acc0 = __builtin_amdgcn_wmma_f32_16x16x32_f16(false, a.v, false, b0.v,
                                                          (short)0, acc0, false, false);
            acc1 = __builtin_amdgcn_wmma_f32_16x16x32_f16(false, a.v, false, b1.v,
                                                          (short)0, acc1, false, false);
        }
    }
    // ---- store C (ISA C/D layout: VGPR r -> M = r + 8*lhi, N = lsub) -----
#pragma unroll
    for (int r = 0; r < 8; ++r) {
        const size_t mrow = (size_t)(m0 + mtile * 16 + r + 8 * lhi);
        C[mrow * 64 + (ntb    ) * 16 + lsub] = acc0[r];
        C[mrow * 64 + (ntb + 1) * 16 + lsub] = acc1[r];
    }
}

// transpose [rows x 64] f32 -> [64 x rows] f16 ; grid = (rows/256, 64)
__global__ void transpose_h16_kernel(const float* __restrict__ X,
                                     h16* __restrict__ XT, int rows)
{
    int c = blockIdx.y;
    int r = blockIdx.x * 256 + threadIdx.x;
    XT[(size_t)c * rows + r] = (h16)X[(size_t)r * 64 + c];
}

// xpre[n][c] = c<32 ? pd_y[n].th_w[c]+b : relu(pd_y[n].thr_w[c-32]+br)
__global__ void x_pre_kernel(const float* __restrict__ pdy,
                             const float* __restrict__ w0, const float* __restrict__ b0,
                             const float* __restrict__ wr, const float* __restrict__ br,
                             float* __restrict__ xpre)
{
    int idx = blockIdx.x * 256 + threadIdx.x;
    int n = idx >> 6, c = idx & 63;
    const float* rowp = pdy + (size_t)n * 64;
    const float* wv; float bias; bool relu;
    if (c < 32) { wv = w0 + c * 64;        bias = b0[c];      relu = false; }
    else        { wv = wr + (c - 32) * 64; bias = br[c - 32]; relu = true;  }
    float s = bias;
#pragma unroll
    for (int k = 0; k < 64; ++k) s += rowp[k] * wv[k];
    if (relu && s < 0.f) s = 0.f;
    xpre[idx] = s;
}

// per-column sum / sumsq over `rows`; grid = 64 (one block per column)
__global__ void col_stats_kernel(const float* __restrict__ X, int rows,
                                 float* __restrict__ stats)
{
    int c = blockIdx.x, t = threadIdx.x;
    float s = 0.f, s2 = 0.f;
    for (int r = t; r < rows; r += 256) {
        float v = X[(size_t)r * 64 + c];
        s += v; s2 += v * v;
    }
    __shared__ float sh[256], sh2[256];
    sh[t] = s; sh2[t] = s2;
    __syncthreads();
    for (int ofs = 128; ofs > 0; ofs >>= 1) {
        if (t < ofs) { sh[t] += sh[t + ofs]; sh2[t] += sh2[t + ofs]; }
        __syncthreads();
    }
    if (t == 0) { stats[c] = sh[0]; stats[64 + c] = sh2[0]; }
}

// BN on xpre, emit transposed f16 xT16[64][4096]; grid = (16, 64)
__global__ void bn_x_kernel(const float* __restrict__ xpre, const float* __restrict__ stats,
                            const float* __restrict__ w, const float* __restrict__ b,
                            h16* __restrict__ xT)
{
    int c = blockIdx.y;
    int n = blockIdx.x * 256 + threadIdx.x;
    const float inv = 1.0f / (float)NN;
    float m  = stats[c] * inv;
    float v  = stats[64 + c] * inv - m * m;
    float sc = rsqrtf(v + 1e-5f) * w[c];
    xT[(size_t)c * NN + n] = (h16)((xpre[(size_t)n * 64 + c] - m) * sc + b[c]);
}

// ypre = concat(lg_y@Ga.T+ba + pm_x@Gx.T+bx , relu(lg_y@Gar.T+bar + pm_x@Gxr.T+bxr))
__global__ void y_pre_kernel(const float* __restrict__ lgy, const float* __restrict__ pmx,
                             const float* __restrict__ gaw, const float* __restrict__ gab,
                             const float* __restrict__ gxw, const float* __restrict__ gxb,
                             const float* __restrict__ garw, const float* __restrict__ garb,
                             const float* __restrict__ gxrw, const float* __restrict__ gxrb,
                             float* __restrict__ ypre)
{
    int idx = blockIdx.x * 256 + threadIdx.x;
    int e = idx >> 6, c = idx & 63;
    const float* ra = lgy + (size_t)e * 64;
    const float* rx = pmx + (size_t)e * 64;
    const float *wa, *wx; float bias; bool relu;
    if (c < 32) { wa = gaw  + c * 64;        wx = gxw  + c * 64;
                  bias = gab[c] + gxb[c];    relu = false; }
    else        { wa = garw + (c - 32) * 64; wx = gxrw + (c - 32) * 64;
                  bias = garb[c - 32] + gxrb[c - 32]; relu = true; }
    float s = bias;
#pragma unroll
    for (int k = 0; k < 64; ++k) s += ra[k] * wa[k] + rx[k] * wx[k];
    if (relu && s < 0.f) s = 0.f;
    ypre[idx] = s;
}

// BN on ypre -> d_out f32
__global__ void bn_y_kernel(const float* __restrict__ ypre, const float* __restrict__ stats,
                            const float* __restrict__ w, const float* __restrict__ b,
                            float* __restrict__ out)
{
    int idx = blockIdx.x * 256 + threadIdx.x;
    int c = idx & 63;
    const float inv = 1.0f / (float)EE;
    float m = stats[c] * inv;
    float v = stats[64 + c] * inv - m * m;
    out[idx] = (ypre[idx] - m) * rsqrtf(v + 1e-5f) * w[c] + b[c];
}

extern "C" void kernel_launch(void* const* d_in, const int* in_sizes, int n_in,
                              void* d_out, int out_size, void* d_ws, size_t ws_size,
                              hipStream_t stream)
{
    const float* y     = (const float*)d_in[0];
    const float* lg_a1 = (const float*)d_in[3];
    const float* pm    = (const float*)d_in[4];
    const float* pd    = (const float*)d_in[5];
    const float* th_w  = (const float*)d_in[6];
    const float* th_b  = (const float*)d_in[7];
    const float* thr_w = (const float*)d_in[8];
    const float* thr_b = (const float*)d_in[9];
    const float* ga_w  = (const float*)d_in[10];
    const float* ga_b  = (const float*)d_in[11];
    const float* gx_w  = (const float*)d_in[12];
    const float* gx_b  = (const float*)d_in[13];
    const float* gar_w = (const float*)d_in[14];
    const float* gar_b = (const float*)d_in[15];
    const float* gxr_w = (const float*)d_in[16];
    const float* gxr_b = (const float*)d_in[17];
    const float* bnx_w = (const float*)d_in[18];
    const float* bnx_b = (const float*)d_in[19];
    const float* bny_w = (const float*)d_in[20];
    const float* bny_b = (const float*)d_in[21];
    float* out = (float*)d_out;

    char* ws = (char*)d_ws;
    h16*   yT16   = (h16*)(ws);                        // 64*8192*2 = 1 MiB
    h16*   xT16   = (h16*)(ws + 1048576);              // 64*4096*2 = 512 KiB
    float* pd_y   = (float*)(ws + 1572864);            // 4096*64*4 = 1 MiB
    float* pm_x   = (float*)(ws + 2621440);            // 8192*64*4 = 2 MiB
    float* lg_y   = (float*)(ws + 4718592);            // 2 MiB
    float* xpre   = (float*)(ws + 6815744);            // 1 MiB
    float* ypre   = (float*)(ws + 7864320);            // 2 MiB
    float* statsX = (float*)(ws + 9961472);            // 128 f32
    float* statsY = (float*)(ws + 9961984);            // 128 f32

    // 1) y -> yT16
    transpose_h16_kernel<<<dim3(EE / 256, 64), 256, 0, stream>>>(y, yT16, EE);
    // 2) pd_y = pd @ y            [4096 x 8192] * [8192 x 64]
    gemm_wmma_kernel<false><<<NN / 64, 256, 0, stream>>>(pd, yT16, pd_y, NN, EE, EE);
    // 3) theta linears + concat/relu
    x_pre_kernel<<<(NN * 64) / 256, 256, 0, stream>>>(pd_y, th_w, th_b, thr_w, thr_b, xpre);
    // 4) BN-x stats + apply (emit f16 transposed x)
    col_stats_kernel<<<64, 256, 0, stream>>>(xpre, NN, statsX);
    bn_x_kernel<<<dim3(NN / 256, 64), 256, 0, stream>>>(xpre, statsX, bnx_w, bnx_b, xT16);
    // 5) pm_x = pm.T @ x          [8192 x 4096] * [4096 x 64] (A column-major source)
    gemm_wmma_kernel<true><<<EE / 64, 256, 0, stream>>>(pm, xT16, pm_x, EE, NN, EE);
    // 6) lg_y = lg_a1 @ y         [8192 x 8192] * [8192 x 64]
    gemm_wmma_kernel<false><<<EE / 64, 256, 0, stream>>>(lg_a1, yT16, lg_y, EE, EE, EE);
    // 7) gamma linears + concat/relu
    y_pre_kernel<<<(EE * 64) / 256, 256, 0, stream>>>(lg_y, pm_x, ga_w, ga_b, gx_w, gx_b,
                                                      gar_w, gar_b, gxr_w, gxr_b, ypre);
    // 8) BN-y stats + apply -> out
    col_stats_kernel<<<64, 256, 0, stream>>>(ypre, EE, statsY);
    bn_y_kernel<<<(EE * 64) / 256, 256, 0, stream>>>(ypre, statsY, bny_w, bny_b, out);
}
